// GraphConvolution_41034117546037
// MI455X (gfx1250) — matrix operand order repaired and checked by reading, not compiled, not run on hardware
//
#include <hip/hip_runtime.h>

typedef __attribute__((ext_vector_type(16))) __bf16 v16bf;
typedef __attribute__((ext_vector_type(8)))  float  v8f;

#define N_NODES 50000
#define R_REL   16
#define DIN     128
#define DOUT    128

// f32 -> bf16 round-to-nearest-even, and back (bit ops; no header dependence)
static __device__ __forceinline__ unsigned short f2bf(float x) {
  unsigned u = __float_as_uint(x);
  unsigned r = u + 0x7FFFu + ((u >> 16) & 1u);
  return (unsigned short)(r >> 16);
}
static __device__ __forceinline__ float bf2f(unsigned short h) {
  return __uint_as_float(((unsigned)h) << 16);
}

// ---------------------------------------------------------------------------
// K1: W_F[r] = sum_b comp[r,b] * basis[b], split into bf16 hi/lo and written
// pre-swizzled into this code's contiguous-k per-lane B-fragment order for
// v_wmma_f32_16x16x32_bf16:  B(32x16): lane = 16*(K/16) + n, VGPR v holds
// bf16 pair (K0+2v, K0+2v+1) with K0 = 16*(lane/16).
// Lane-contiguous layout so k_rgcn loads 2x global_load_b128 per fragment:
//   bf[ ((((r*8 + w)*4 + s)*32 + lane)*8 + v) ]
// ---------------------------------------------------------------------------
__global__ void k_build_wf(const float* __restrict__ wF, const float* __restrict__ comp,
                           unsigned* __restrict__ bfHi, unsigned* __restrict__ bfLo) {
  int idx = blockIdx.x * 256 + threadIdx.x;   // 16 * 64 * 128 = 131072 threads
  int r   = idx >> 13;                        // / (64*128)
  int rem = idx & 8191;
  int jp  = rem >> 7;                         // K-pair index 0..63
  int k   = rem & 127;                        // output column 0..127
  int j0  = jp * 2;
  float w0 = 0.f, w1 = 0.f;
#pragma unroll
  for (int b = 0; b < 8; ++b) {
    float c = comp[r * 8 + b];
    const float* base = wF + ((size_t)b * 128 + j0) * 128 + k;
    w0 = fmaf(c, base[0], w0);
    w1 = fmaf(c, base[128], w1);
  }
  unsigned short h0 = f2bf(w0), h1 = f2bf(w1);
  unsigned short l0 = f2bf(w0 - bf2f(h0)), l1 = f2bf(w1 - bf2f(h1));
  int s = j0 >> 5, kk = j0 & 31, hh = kk >> 4, v = (kk & 15) >> 1;
  int wb = k >> 4, nn = k & 15, lane = hh * 16 + nn;
  int o = ((((r * 8 + wb) * 4 + s) * 32 + lane) * 8 + v);
  bfHi[o] = (unsigned)h0 | ((unsigned)h1 << 16);
  bfLo[o] = (unsigned)l0 | ((unsigned)l1 << 16);
}

// ---------------------------------------------------------------------------
// Counting sort of edges by destination row.
// ---------------------------------------------------------------------------
__global__ void k_zero(int* __restrict__ p, int n) {
  int i = blockIdx.x * 256 + threadIdx.x;
  if (i < n) p[i] = 0;
}

__global__ void k_hist(const int* __restrict__ rows, int* __restrict__ cnt, int nnz) {
  int e = blockIdx.x * 256 + threadIdx.x;
  if (e < nnz) atomicAdd(&cnt[rows[e]], 1);
}

__global__ void k_scan(const int* __restrict__ cnt, int* __restrict__ ptr,
                       int* __restrict__ cur, int n) {
  __shared__ int part[1024];
  int t = threadIdx.x;
  const int CH = (n + 1023) / 1024;           // 49
  int b0 = t * CH;
  int b1 = b0 + CH; if (b1 > n) b1 = n;
  int s = 0;
  for (int i = b0; i < b1; ++i) s += cnt[i];
  part[t] = s;
  __syncthreads();
  for (int off = 1; off < 1024; off <<= 1) {
    int v = (t >= off) ? part[t - off] : 0;
    __syncthreads();
    part[t] += v;
    __syncthreads();
  }
  int run = part[t] - s;                      // exclusive prefix
  for (int i = b0; i < b1; ++i) { ptr[i] = run; cur[i] = run; run += cnt[i]; }
  if (b0 < n && b1 == n) ptr[n] = run;        // total
}

__global__ void k_scatter(const int* __restrict__ rows, const int* __restrict__ cols,
                          const float* __restrict__ vals, int* __restrict__ cur,
                          uint2* __restrict__ recs, int nnz) {
  int e = blockIdx.x * 256 + threadIdx.x;
  if (e >= nnz) return;
  int row = rows[e];
  int col = cols[e];
  int r = col / N_NODES;
  int n = col - r * N_NODES;                  // n < 50000 < 65536 -> fits 16 bits
  int pos = atomicAdd(&cur[row], 1);
  recs[pos] = make_uint2(((unsigned)r << 16) | (unsigned)n, __float_as_uint(vals[e]));
}

// ---------------------------------------------------------------------------
// Main kernel: one workgroup (256 thr = 8 waves) owns 16 destination rows.
// Phase 1: gather val*X[n] (L2-resident) into LDS agg[r][m][128] (f32),
//          wave w exclusively owns rows {2w, 2w+1}  -> no atomics anywhere.
// Phase 2: in-place repack f32 -> (bf16_hi | bf16_lo<<16) per dword.
// Phase 3: out[16g+M][:] = sum_r agg_r @ W_F[r] via bf16 WMMA with hi/lo
//          error compensation (hi*hi + hi*lo + lo*hi), f32 accumulate.
// ---------------------------------------------------------------------------
__global__ void __launch_bounds__(256) k_rgcn(
    const float* __restrict__ X, const int* __restrict__ ptr,
    const uint2* __restrict__ recs,
    const unsigned* __restrict__ bfHi, const unsigned* __restrict__ bfLo,
    float* __restrict__ out) {
  extern __shared__ float agg[];              // [16r][16m][128] f32 = 128 KB
  const int tid  = threadIdx.x;
  const int wave = tid >> 5, lane = tid & 31;
  const int g    = blockIdx.x;                // rows 16g .. 16g+15

  for (int i = tid; i < R_REL * 16 * DIN; i += 256) agg[i] = 0.0f;
  __syncthreads();

  // ---- Phase 1: aggregate neighbors ----
#pragma unroll
  for (int mi = 0; mi < 2; ++mi) {
    int m   = wave * 2 + mi;
    int row = g * 16 + m;
    int e0 = ptr[row], e1 = ptr[row + 1];
    for (int e = e0; e < e1; ++e) {
      if (e + 8 < e1) __builtin_prefetch(&recs[e + 8], 0, 0);
      uint2 rec = recs[e];
      int   r   = rec.x >> 16;
      int   n   = rec.x & 0xFFFF;
      float val = __uint_as_float(rec.y);
      float4 xv = ((const float4*)(X + (size_t)n * DIN))[lane];   // global_load_b128
      float4* dp = ((float4*)(agg + (r * 16 + m) * DIN)) + lane;  // ds b128 RMW
      float4 a = *dp;
      a.x = fmaf(val, xv.x, a.x);
      a.y = fmaf(val, xv.y, a.y);
      a.z = fmaf(val, xv.z, a.z);
      a.w = fmaf(val, xv.w, a.w);
      *dp = a;
    }
  }
  __syncthreads();

  // ---- Phase 2: in-place f32 -> packed (bf16 hi | bf16 lo << 16) ----
  unsigned* aggu = (unsigned*)agg;
  for (int i = tid; i < R_REL * 16 * DIN; i += 256) {
    float x = agg[i];
    unsigned short h = f2bf(x);
    unsigned short l = f2bf(x - bf2f(h));
    aggu[i] = (unsigned)h | ((unsigned)l << 16);
  }
  __syncthreads();

  // ---- Phase 3: WMMA transform, accumulate over all 16 relations ----
  const int hl = lane >> 4, nl = lane & 15;
  v8f C = {};
  for (int r = 0; r < R_REL; ++r) {
#pragma unroll
    for (int s = 0; s < 4; ++s) {
      // B(32x16 bf16) fragments: per-lane contiguous -> 2x global_load_b128 each
      union { unsigned u[8]; uint4 q[2]; v16bf v; } Bh, Bl;
      {
        size_t base = ((((size_t)r * 8 + wave) * 4 + s) * 32 + lane) * 8;
        const uint4* bh4 = (const uint4*)(bfHi + base);
        const uint4* bl4 = (const uint4*)(bfLo + base);
        Bh.q[0] = bh4[0]; Bh.q[1] = bh4[1];
        Bl.q[0] = bl4[0]; Bl.q[1] = bl4[1];
      }
      // A(16x32 bf16) fragments from LDS: lane m = nl, K = 32s+16hl+2v
      union { unsigned u[8]; v16bf v; } Ah, Al;
#pragma unroll
      for (int vv = 0; vv < 8; ++vv) {
        int g2 = vv >> 2, u2 = vv & 3;
        int K  = 32 * s + 16 * hl + 2 * vv;
        uint2 pk = *(const uint2*)&aggu[(r * 16 + nl) * DIN + K]; // ds_read_b64
        Ah.u[vv] = (pk.x & 0xFFFFu) | (pk.y << 16);
        Al.u[vv] = (pk.x >> 16)     | (pk.y & 0xFFFF0000u);
      }
      // hi*hi + hi*lo + lo*hi : near-f32 precision, f32 accumulation
      C = __builtin_amdgcn_wmma_f32_16x16x32_bf16(false, Ah.v, false, Bh.v, (short)0, C, false, false);
      C = __builtin_amdgcn_wmma_f32_16x16x32_bf16(false, Ah.v, false, Bl.v, (short)0, C, false, false);
      C = __builtin_amdgcn_wmma_f32_16x16x32_bf16(false, Al.v, false, Bh.v, (short)0, C, false, false);
    }
  }

  // ---- Store: D layout lane nl = N col, VGPR j -> M = j + 8*hl.
  //      Exclusive ownership; non-temporal (write-once, keep L2 for X/recs). ----
#pragma unroll
  for (int j = 0; j < 8; ++j) {
    int M = j + 8 * hl;
    __builtin_nontemporal_store(C[j], &out[(size_t)(g * 16 + M) * DOUT + wave * 16 + nl]);
  }
}

// ---------------------------------------------------------------------------
extern "C" void kernel_launch(void* const* d_in, const int* in_sizes, int n_in,
                              void* d_out, int out_size, void* d_ws, size_t ws_size,
                              hipStream_t stream) {
  const float* X    = (const float*)d_in[0];
  const int*   Arow = (const int*)d_in[1];
  const int*   Acol = (const int*)d_in[2];
  const float* Aval = (const float*)d_in[3];
  const float* wF   = (const float*)d_in[4];
  const float* comp = (const float*)d_in[5];
  float* out = (float*)d_out;
  const int nnz = in_sizes[1];

  // Workspace layout (~14.6 MB total)
  char* ws = (char*)d_ws;
  unsigned* bfHi = (unsigned*)(ws);                        // 512 KB swizzled B hi
  unsigned* bfLo = (unsigned*)(ws + (512 << 10));          // 512 KB swizzled B lo
  int*   cnt  = (int*)(ws + (1 << 20));                    // 50000 ints
  int*   ptr  = (int*)(ws + (1 << 20) + (256 << 10));      // 50001 ints
  int*   cur  = (int*)(ws + (1 << 20) + (512 << 10));      // 50000 ints
  uint2* recs = (uint2*)(ws + (1 << 20) + (768 << 10));    // nnz * 8 B

  k_build_wf<<<512, 256, 0, stream>>>(wF, comp, bfHi, bfLo);
  k_zero<<<(N_NODES + 255) / 256, 256, 0, stream>>>(cnt, N_NODES);
  k_hist<<<(nnz + 255) / 256, 256, 0, stream>>>(Arow, cnt, nnz);
  k_scan<<<1, 1024, 0, stream>>>(cnt, ptr, cur, N_NODES);
  k_scatter<<<(nnz + 255) / 256, 256, 0, stream>>>(Arow, Acol, Aval, cur, recs, nnz);
  k_rgcn<<<N_NODES / 16, 256, R_REL * 16 * DIN * 4, stream>>>(X, ptr, recs, bfHi, bfLo, out);
}
